// ReDoTrAS_40561671143561
// MI455X (gfx1250) — compile-verified, matrix operand order checked by reading, not run on hardware
//
#include <hip/hip_runtime.h>
#include <hip/hip_bf16.h>

typedef __attribute__((ext_vector_type(16))) _Float16 v16h;
typedef __attribute__((ext_vector_type(8)))  float    v8f;

#define WMMA16(a, b, c) \
  __builtin_amdgcn_wmma_f32_16x16x32_f16(false, (a), false, (b), (short)0, (c), false, false)

static constexpr int Bb   = 2;
static constexpr int Ss   = 4096;
static constexpr int Dd   = 1024;
static constexpr int SEGc = 512;
static constexpr int STc  = 64;
static constexpr int Hh   = 16;
static constexpr int DKc  = 64;
static constexpr int DVc  = 64;
static constexpr int Lc   = STc + SEGc + STc;   // 640
static constexpr int NSEG = Ss / SEGc;          // 8

// ---------------- fragment loaders (CDNA5 WMMA VGPR layouts, wave32) -------

// A-matrix 16x32 f16, row-major source (lda in elements).
// lane<16: rows M=lane, K = 0..7 & 16..23 ; lane>=16: same rows, K = 8..15 & 24..31
__device__ inline v16h load_a_frag(const _Float16* __restrict__ A, int lda, int m0, int k0) {
  int lane = threadIdx.x & 31;
  const _Float16* p = A + (size_t)(m0 + (lane & 15)) * lda + k0 + ((lane >> 4) << 3);
  v16h a;
#pragma unroll
  for (int i = 0; i < 8; ++i) { a[i] = p[i]; a[8 + i] = p[16 + i]; }
  return a;
}

// B-matrix 32x16 fragment from a *transposed* (N x K row-major) source.
// Per-lane accesses are two contiguous 16-byte chunks -> global_load_b128.
__device__ inline v16h load_bT_frag(const _Float16* __restrict__ Bt, int ldbt, int k0, int n0) {
  int lane = threadIdx.x & 31;
  const _Float16* p = Bt + (size_t)(n0 + (lane & 15)) * ldbt + k0 + ((lane >> 4) << 3);
  v16h b;
#pragma unroll
  for (int i = 0; i < 8; ++i) { b[i] = p[i]; b[8 + i] = p[16 + i]; }
  return b;
}

// ---------------- fp32 -> f16 conversion kernels ---------------------------

__global__ void cvt_f32_f16_kernel(const float* __restrict__ src, _Float16* __restrict__ dst,
                                   size_t n) {
  size_t i = (size_t)blockIdx.x * blockDim.x + threadIdx.x;
  size_t stride = (size_t)gridDim.x * blockDim.x;
  for (; i < n; i += stride) dst[i] = (_Float16)src[i];
}

// src: (batch, rows, cols) f32 -> dst: (batch, cols, rows) f16 ; write-contiguous
__global__ void cvt_transpose_kernel(const float* __restrict__ src, _Float16* __restrict__ dst,
                                     int rows, int cols) {
  size_t slab = (size_t)rows * cols;
  const float* s = src + (size_t)blockIdx.y * slab;
  _Float16* d = dst + (size_t)blockIdx.y * slab;
  size_t i = (size_t)blockIdx.x * blockDim.x + threadIdx.x;
  size_t stride = (size_t)gridDim.x * blockDim.x;
  for (; i < slab; i += stride) {
    size_t c = i / rows;   // output row  = source column
    size_t r = i % rows;   // output col  = source row
    d[i] = (_Float16)s[r * cols + c];
  }
}

// ---------------- QKV projection + fused RoPE ------------------------------
// Weights are pre-transposed per head: WT[h] is DK x D row-major.

struct QKVW { const _Float16* w[3][3]; };  // [mat: q,k,v][region: ss,mid,se]

__global__ void qkv_rope_kernel(const _Float16* __restrict__ xH,
                                const _Float16* __restrict__ stH, QKVW wts,
                                _Float16* __restrict__ qH, _Float16* __restrict__ kH,
                                _Float16* __restrict__ vTH, int seg) {
  int wid = blockIdx.x * (blockDim.x >> 5) + (threadIdx.x >> 5);
  int mat = wid / 5120;            // 0=q 1=k 2=v
  int rem = wid % 5120;
  int b = rem / 2560;  rem %= 2560;
  int h = rem / 160;   rem %= 160;
  int mt = rem / 4;                // L tile (40 tiles of 16)
  int nt = rem % 4;                // DK tile

  const _Float16* Asrc;
  int m0, region;
  if (mt < 4)       { Asrc = stH + (size_t)b * STc * Dd;                      m0 = mt * 16;        region = 0; }
  else if (mt < 36) { Asrc = xH + ((size_t)b * Ss + (size_t)seg * SEGc) * Dd; m0 = (mt - 4) * 16;  region = 1; }
  else              { Asrc = stH + (size_t)b * STc * Dd;                      m0 = (mt - 36) * 16; region = 2; }
  const _Float16* WT = wts.w[mat][region] + (size_t)h * DKc * Dd;  // DK x D

  v8f acc = {};
  for (int k0 = 0; k0 < Dd; k0 += 32) {
    v16h a  = load_a_frag(Asrc, Dd, m0, k0);
    v16h bf = load_bT_frag(WT, Dd, k0, nt * 16);
    acc = WMMA16(a, bf, acc);
  }

  int lane = threadIdx.x & 31;
  int n = lane & 15, mb = (lane >> 4) << 3;
  int ncol = nt * 16 + n;

  if (mat < 2) {
    // rope: inv_freq = 10000^(-(2i)/64); even/odd column pairing via lane xor 1
    _Float16* dst = (mat == 0 ? qH : kH) + (size_t)(b * Hh + h) * Lc * DKc;
    float invf = __expf(-0.28782313662425575f * (float)(ncol >> 1));
#pragma unroll
    for (int r = 0; r < 8; ++r) {
      float v    = acc[r];
      float part = __shfl_xor(v, 1, 32);
      int Lrow   = mt * 16 + mb + r;
      float pos  = (float)(seg * SEGc - STc + Lrow);
      float ang  = pos * invf;
      float cs = __cosf(ang), sn = __sinf(ang);
      float res = (ncol & 1) ? (part * sn + v * cs) : (v * cs - part * sn);
      dst[(size_t)Lrow * DKc + ncol] = (_Float16)res;
    }
  } else {
    // V stored transposed: [b][h][dv][L] so attention's B-fragments are contiguous
    _Float16* dst = vTH + (size_t)(b * Hh + h) * DVc * Lc + (size_t)ncol * Lc;
#pragma unroll
    for (int r = 0; r < 8; ++r) {
      int Lrow = mt * 16 + mb + r;
      dst[Lrow] = (_Float16)acc[r];
    }
  }
}

// ---------------- flash attention, one wave per 16-row q tile --------------

__global__ void attn_kernel(const _Float16* __restrict__ qH, const _Float16* __restrict__ kH,
                            const _Float16* __restrict__ vTH, _Float16* __restrict__ attH) {
  __shared__ _Float16 Pt[8][16 * 32];  // per-wave P tile staging (C-layout -> A-layout)
  int wslot = threadIdx.x >> 5;
  int wid = blockIdx.x * 8 + wslot;
  int b = wid / 640;
  int rem = wid % 640;
  int h = rem / 40;
  int qt = rem % 40;
  int qm0 = qt * 16;
  int lane = threadIdx.x & 31;
  int n = lane & 15, mb = (lane >> 4) << 3;

  const _Float16* qb = qH + (size_t)(b * Hh + h) * Lc * DKc;
  const _Float16* kb = kH + (size_t)(b * Hh + h) * Lc * DKc;
  const _Float16* vb = vTH + (size_t)(b * Hh + h) * DVc * Lc;  // DV x L

  v16h a0 = load_a_frag(qb, DKc, qm0, 0);
  v16h a1 = load_a_frag(qb, DKc, qm0, 32);

  v8f o0 = {}, o1 = {}, o2 = {}, o3 = {};
  float mrow[8], lrow[8];
#pragma unroll
  for (int r = 0; r < 8; ++r) { mrow[r] = -3.0e38f; lrow[r] = 0.f; }

  int kend = qm0 + 15;  // causal: last needed key
  for (int kc = 0; kc <= kend; kc += 32) {
    v8f s0 = {}, s1 = {};
    s0 = WMMA16(a0, load_bT_frag(kb, DKc, 0, kc), s0);       // k is naturally N x K
    s0 = WMMA16(a1, load_bT_frag(kb, DKc, 32, kc), s0);
    s1 = WMMA16(a0, load_bT_frag(kb, DKc, 0, kc + 16), s1);
    s1 = WMMA16(a1, load_bT_frag(kb, DKc, 32, kc + 16), s1);

    int c0 = kc + n, c1 = kc + 16 + n;
    float al[8];
#pragma unroll
    for (int r = 0; r < 8; ++r) {
      int mg = qm0 + mb + r;
      float x0 = (c0 > mg) ? -3.0e38f : s0[r] * 0.125f;
      float x1 = (c1 > mg) ? -3.0e38f : s1[r] * 0.125f;
      float tm = fmaxf(x0, x1);
      for (int msk = 1; msk < 16; msk <<= 1) tm = fmaxf(tm, __shfl_xor(tm, msk, 32));
      float newm  = fmaxf(mrow[r], tm);
      float alpha = __expf(mrow[r] - newm);
      float p0 = __expf(x0 - newm);
      float p1 = __expf(x1 - newm);
      float ts = p0 + p1;
      for (int msk = 1; msk < 16; msk <<= 1) ts += __shfl_xor(ts, msk, 32);
      lrow[r] = lrow[r] * alpha + ts;
      mrow[r] = newm;
      al[r]   = alpha;
      Pt[wslot][(mb + r) * 32 + n]      = (_Float16)p0;
      Pt[wslot][(mb + r) * 32 + 16 + n] = (_Float16)p1;
    }
#pragma unroll
    for (int r = 0; r < 8; ++r) { o0[r] *= al[r]; o1[r] *= al[r]; o2[r] *= al[r]; o3[r] *= al[r]; }

    v16h pa = load_a_frag(&Pt[wslot][0], 32, 0, 0);  // DS ops in-order per wave
    o0 = WMMA16(pa, load_bT_frag(vb, Lc, kc, 0),  o0);
    o1 = WMMA16(pa, load_bT_frag(vb, Lc, kc, 16), o1);
    o2 = WMMA16(pa, load_bT_frag(vb, Lc, kc, 32), o2);
    o3 = WMMA16(pa, load_bT_frag(vb, Lc, kc, 48), o3);
  }

  _Float16* ob = attH + (size_t)b * Lc * (Hh * DVc);
#pragma unroll
  for (int r = 0; r < 8; ++r) {
    float inv = 1.0f / lrow[r];
    size_t base = (size_t)(qm0 + mb + r) * (Hh * DVc) + h * DVc;
    ob[base + n]      = (_Float16)(o0[r] * inv);
    ob[base + 16 + n] = (_Float16)(o1[r] * inv);
    ob[base + 32 + n] = (_Float16)(o2[r] * inv);
    ob[base + 48 + n] = (_Float16)(o3[r] * inv);
  }
}

// ---------------- batched WMMA GEMM (f16 in, f32 out, optional f16 out) ----
// B is pre-transposed: N x K row-major, ldbt == K.

__global__ void gemm_f16_kernel(const _Float16* __restrict__ A, long long sAb, int lda,
                                const _Float16* __restrict__ BmT, int K,
                                float* __restrict__ Cf, long long sCb, int ldc,
                                _Float16* __restrict__ Ch, long long sChb, int ldch,
                                int tilesM, int tilesN) {
  int wid = blockIdx.x * (blockDim.x >> 5) + (threadIdx.x >> 5);
  int perBatch = tilesM * tilesN;
  int batch = wid / perBatch;
  int rem = wid % perBatch;
  int mt = rem / tilesN, nt = rem % tilesN;
  const _Float16* Ab = A + (size_t)batch * sAb;

  int lane = threadIdx.x & 31;
  v8f acc = {};
  for (int k0 = 0; k0 < K; k0 += 32) {
    if (k0 + 64 < K) {  // prefetch next B chunk -> global_prefetch_b8
      __builtin_prefetch(BmT + (size_t)(nt * 16 + (lane & 15)) * K + k0 + 64, 0, 0);
    }
    v16h a  = load_a_frag(Ab, lda, mt * 16, k0);
    v16h bf = load_bT_frag(BmT, K, k0, nt * 16);
    acc = WMMA16(a, bf, acc);
  }

  int n = lane & 15, mb = (lane >> 4) << 3;
  float* cf = Cf + (size_t)batch * sCb;
#pragma unroll
  for (int r = 0; r < 8; ++r)
    cf[(size_t)(mt * 16 + mb + r) * ldc + nt * 16 + n] = acc[r];
  if (Ch) {
    _Float16* ch = Ch + (size_t)batch * sChb;
#pragma unroll
    for (int r = 0; r < 8; ++r)
      ch[(size_t)(mt * 16 + mb + r) * ldch + nt * 16 + n] = (_Float16)acc[r];
  }
}

// ---------------- host orchestration ---------------------------------------

extern "C" void kernel_launch(void* const* d_in, const int* in_sizes, int n_in,
                              void* d_out, int out_size, void* d_ws, size_t ws_size,
                              hipStream_t stream) {
  (void)in_sizes; (void)n_in; (void)out_size; (void)ws_size;
  const float* x  = (const float*)d_in[0];
  const float* st = (const float*)d_in[1];
  float* out = (float*)d_out;

  _Float16* p = (_Float16*)d_ws;
  _Float16* xH = p;   p += (size_t)Bb * Ss * Dd;
  _Float16* wH[9];                                  // transposed: [H][DK][D]
  for (int i = 0; i < 9; ++i) { wH[i] = p; p += (size_t)Hh * Dd * DKc; }
  _Float16* WoT  = p; p += (size_t)Dd * (Hh * DVc); // transposed: [D][H*DV]->N x K
  _Float16* WoST = p; p += (size_t)Dd * (Hh * DVc);
  _Float16* stH  = p; p += (size_t)Bb * STc * Dd;
  _Float16* qH   = p; p += (size_t)Bb * Hh * Lc * DKc;
  _Float16* kH   = p; p += (size_t)Bb * Hh * Lc * DKc;
  _Float16* vTH  = p; p += (size_t)Bb * Hh * DVc * Lc;
  _Float16* attH = p; p += (size_t)Bb * Lc * (Hh * DVc);

  // plain conversions (x, state)
  cvt_f32_f16_kernel<<<1024, 256, 0, stream>>>(x, xH, (size_t)Bb * Ss * Dd);
  cvt_f32_f16_kernel<<<64, 256, 0, stream>>>(st, stH, (size_t)STc * Dd);
  cvt_f32_f16_kernel<<<64, 256, 0, stream>>>(st, stH + (size_t)STc * Dd, (size_t)STc * Dd);

  // transposed weight conversions: per-head D x DK -> DK x D (batch over heads)
  for (int i = 0; i < 9; ++i)
    cvt_transpose_kernel<<<dim3(64, Hh), 256, 0, stream>>>(
        (const float*)d_in[2 + i], wH[i], Dd, DKc);
  // Wo, Wo_state: (H*DV) x D -> D x (H*DV)
  cvt_transpose_kernel<<<dim3(512, 1), 256, 0, stream>>>((const float*)d_in[11], WoT,
                                                         Hh * DVc, Dd);
  cvt_transpose_kernel<<<dim3(512, 1), 256, 0, stream>>>((const float*)d_in[12], WoST,
                                                         Hh * DVc, Dd);

  QKVW wt;
  wt.w[0][0] = wH[4]; wt.w[0][1] = wH[1]; wt.w[0][2] = wH[7];  // q: ss, mid, se
  wt.w[1][0] = wH[3]; wt.w[1][1] = wH[0]; wt.w[1][2] = wH[6];  // k
  wt.w[2][0] = wH[5]; wt.w[2][1] = wH[2]; wt.w[2][2] = wH[8];  // v

  float* outState = out + (size_t)Bb * Ss * Dd;

  for (int seg = 0; seg < NSEG; ++seg) {
    // 3 mats * B*H*40*4 tiles = 15360 waves -> 1920 blocks of 8 waves
    qkv_rope_kernel<<<1920, 256, 0, stream>>>(xH, stH, wt, qH, kH, vTH, seg);
    // B*H*40 q-tiles = 1280 waves -> 160 blocks
    attn_kernel<<<160, 256, 0, stream>>>(qH, kH, vTH, attH);
    // out-proj: 2 batches * 32*64 tiles = 4096 waves -> 512 blocks
    gemm_f16_kernel<<<512, 256, 0, stream>>>(
        attH + (size_t)STc * Dd, (long long)Lc * Dd, Dd, WoT, Dd,
        out + (size_t)seg * SEGc * Dd, (long long)Ss * Dd, Dd,
        (_Float16*)nullptr, 0, 0, SEGc / 16, Dd / 16);
    // state-proj: 2 batches * 4*64 tiles = 512 waves -> 64 blocks
    gemm_f16_kernel<<<64, 256, 0, stream>>>(
        attH + (size_t)(STc + SEGc) * Dd, (long long)Lc * Dd, Dd, WoST, Dd,
        outState, (long long)STc * Dd, Dd,
        stH, (long long)STc * Dd, Dd, STc / 16, Dd / 16);
  }
}
